// PatchReg_71064528879874
// MI455X (gfx1250) — matrix operand order repaired and checked by reading, not compiled, not run on hardware
//
#include <hip/hip_runtime.h>
#include <hip/hip_bf16.h>

// ---------------------------------------------------------------------------
// PatchReg for MI455X (gfx1250, wave32, WMMA).
//  x(8,64,224,224) f32 -> out(8,768,56,56) f32
//  Heavy math routed through v_wmma_f32_16x16x32_f16 (f16 in, f32 acc).
//
// Fragment layouts (CDNA5 ISA 7.12.2):
//  A 16x32 f16 : lane(M=lane&15); lanes<16 hold k {0..7}U{16..23},
//                lanes>=16 hold k {8..15}U{24..31}   (interleaved halves)
//  B 32x16 f16 : lane(N=lane&15); lanes<16 hold k {0..7}U{16..23},
//                lanes>=16 hold k {8..15}U{24..31}   (interleaved halves)
//  C/D 16x16 f32: VGPR r -> row r (lanes<16) / r+8 (lanes>=16), col=lane&15
// ---------------------------------------------------------------------------

typedef __attribute__((ext_vector_type(16))) _Float16 v16h;
typedef __attribute__((ext_vector_type(8)))  _Float16 v8h;
typedef __attribute__((ext_vector_type(8)))  float    v8f;
typedef __attribute__((ext_vector_type(4)))  float    v4f;

#define BB   8
#define CC   64
#define HH   224
#define WW   224
#define EE   768
#define PP   4
#define PH   56
#define PW   56
#define HWPX (HH*WW)          // 50176 pixels per image
#define PPI  (PH*PW)          // 3136 patches per image
#define NPATCH (BB*PPI)       // 25088
#define KTOT 1024             // C * p * p

// ---------------------------------------------------------------------------
// Weight repack: f32 -> f16, w2 re-indexed to k = (dy*4+dx)*64 + c
// ---------------------------------------------------------------------------
__global__ void repack_w1_kernel(const float* __restrict__ w1,
                                 _Float16* __restrict__ w1p) {
  int t = blockIdx.x * 256 + threadIdx.x;
  if (t < CC * CC) w1p[t] = (_Float16)w1[t];        // [cout][cin], cin contig
}

__global__ void repack_w2_kernel(const float* __restrict__ w2,
                                 _Float16* __restrict__ w2p) {
  int t = blockIdx.x * 256 + threadIdx.x;
  if (t < EE * KTOT) {
    int e = t >> 10, k = t & 1023;
    int pix = k >> 6, c = k & 63;                   // pix = dy*4+dx
    // w2 layout (E,C,4,4): idx = e*1024 + c*16 + pix
    w2p[t] = (_Float16)w2[e * 1024 + c * 16 + pix];
  }
}

// ---------------------------------------------------------------------------
// Grid kernel: one wave32 per patch.
//   pooled = leaky(avgpool4(x)); offs/scales = 1x1 conv (2 outs);
//   coords = patch_ref + self*(1+scale) + off  -> grid (B,H,W,2) f32
// ---------------------------------------------------------------------------
__global__ void patch_grid_kernel(const float* __restrict__ x,
                                  const float* __restrict__ w_off,
                                  const float* __restrict__ b_off,
                                  const float* __restrict__ w_sc,
                                  const float* __restrict__ b_sc,
                                  float* __restrict__ grid) {
  const int wave = threadIdx.x >> 5;
  const int lane = threadIdx.x & 31;
  const int p = blockIdx.x * 8 + wave;              // patch id, exact cover
  const int b  = p / PPI;
  const int rem = p - b * PPI;
  const int ph = rem / PW;
  const int pw = rem - ph * PW;

  float dox = 0.f, doy = 0.f, dsx = 0.f, dsy = 0.f;
#pragma unroll
  for (int j = 0; j < 2; ++j) {
    const int c = lane * 2 + j;
    const float* base =
        x + ((size_t)(b * CC + c) * HH + ph * PP) * WW + pw * PP;
    v4f r0 = *(const v4f*)(base + 0 * WW);
    v4f r1 = *(const v4f*)(base + 1 * WW);
    v4f r2 = *(const v4f*)(base + 2 * WW);
    v4f r3 = *(const v4f*)(base + 3 * WW);
    v4f s4 = r0 + r1 + r2 + r3;
    float avg = (s4[0] + s4[1] + s4[2] + s4[3]) * 0.0625f;
    float v = avg > 0.f ? avg : 0.01f * avg;        // leaky relu
    dox += v * w_off[c];      doy += v * w_off[CC + c];
    dsx += v * w_sc[c];       dsy += v * w_sc[CC + c];
  }
  // wave32 xor reduction -> every lane has the full sums
#pragma unroll
  for (int off = 16; off > 0; off >>= 1) {
    dox += __shfl_xor(dox, off, 32);
    doy += __shfl_xor(doy, off, 32);
    dsx += __shfl_xor(dsx, off, 32);
    dsy += __shfl_xor(dsy, off, 32);
  }
  const float offx = (dox + b_off[0]) * (1.f / 56.f);
  const float offy = (doy + b_off[1]) * (1.f / 56.f);
  const float scx  = dsx + b_sc[0];
  const float scy  = dsy + b_sc[1];

  if (lane < 16) {
    const int dy = lane >> 2, dx = lane & 3;
    const int h = ph * PP + dy, w = pw * PP + dx;
    const float prx = -1.f + (2.f / 223.f) * ((float)(pw * PP) + 1.5f);
    const float pry = -1.f + (2.f / 223.f) * ((float)(ph * PP) + 1.5f);
    const float ewx = ((float)dx - 1.5f) * (2.f / 223.f);
    const float ehy = ((float)dy - 1.5f) * (2.f / 223.f);
    const float gx = prx + ewx * (1.f + scx) + offx;
    const float gy = pry + ehy * (1.f + scy) + offy;
    size_t gi = ((size_t)(b * HH + h) * WW + w) * 2;
    grid[gi + 0] = gx;
    grid[gi + 1] = gy;
  }
}

// ---------------------------------------------------------------------------
// Fragment loaders (row base must have k contiguous; 16B-aligned chunks).
// ---------------------------------------------------------------------------
template <typename PtrT>
__device__ __forceinline__ v16h load_fragA_k32(PtrT rowBase, int kbase,
                                               int laneHi) {
  // interleaved halves: {0..7, 16..23} / {8..15, 24..31}
  const _Float16* p0 = (const _Float16*)rowBase + kbase + (laneHi ? 8 : 0);
  v8h lo = *(const v8h*)(p0);
  v8h hi = *(const v8h*)(p0 + 16);
  return __builtin_shufflevector(lo, hi, 0, 1, 2, 3, 4, 5, 6, 7, 8, 9, 10, 11,
                                 12, 13, 14, 15);
}

template <typename PtrT>
__device__ __forceinline__ v16h load_fragB_k32(PtrT rowBase, int kbase,
                                               int laneHi) {
  // interleaved halves: {0..7, 16..23} / {8..15, 24..31}
  const _Float16* p0 = (const _Float16*)rowBase + kbase + (laneHi ? 8 : 0);
  v8h lo = *(const v8h*)(p0);
  v8h hi = *(const v8h*)(p0 + 16);
  return __builtin_shufflevector(lo, hi, 0, 1, 2, 3, 4, 5, 6, 7, 8, 9, 10, 11,
                                 12, 13, 14, 15);
}

// ---------------------------------------------------------------------------
// Tokens kernel: 1x1 conv as GEMM, tokens f16 stored BHWC.
//   block = 256 thr (8 waves); tile = 32 pixels x 64 couts; K = 64 (2 chunks)
// ---------------------------------------------------------------------------
#define TOKROW 72   // LDS row stride (f16), 144B -> 16B aligned, conflict skew

__global__ void tokens_kernel(const float* __restrict__ x,
                              const _Float16* __restrict__ w1p,
                              const float* __restrict__ b1,
                              _Float16* __restrict__ tokens) {
  __shared__ _Float16 Atile[32 * TOKROW];           // [pixel][cin]

  const int b = blockIdx.x / 1568;
  const int pixBase = (blockIdx.x - b * 1568) * 32;
  const int t = threadIdx.x;

  // Phase 1: load 64ch x 32px f32 tile, transpose to LDS as f16 [px][c]
  {
    const int c = t >> 2;
    const int px0 = (t & 3) * 8;
    const float* src = x + (size_t)(b * CC + c) * HWPX + pixBase + px0;
    v4f a = *(const v4f*)(src);
    v4f bvec = *(const v4f*)(src + 4);
#pragma unroll
    for (int i = 0; i < 4; ++i) Atile[(px0 + i) * TOKROW + c] = (_Float16)a[i];
#pragma unroll
    for (int i = 0; i < 4; ++i)
      Atile[(px0 + 4 + i) * TOKROW + c] = (_Float16)bvec[i];
  }
  __syncthreads();

  // Phase 2: 8 waves -> 2 (M) x 4 (N) grid of 16x16 output tiles
  const int wave = t >> 5;
  const int lane = t & 31;
  const int laneHi = lane >= 16;
  const int mSub = wave >> 2;                       // 0..1  (pixel group)
  const int nSub = wave & 3;                        // 0..3  (cout group)

  v8f acc = {};
#pragma unroll
  for (int kc = 0; kc < 2; ++kc) {
    const int kbase = kc * 32;
    v16h afrag = load_fragA_k32(&Atile[(mSub * 16 + (lane & 15)) * TOKROW],
                                kbase, laneHi);
    v16h bfrag = load_fragB_k32(w1p + (size_t)(nSub * 16 + (lane & 15)) * CC,
                                kbase, laneHi);
    acc = __builtin_amdgcn_wmma_f32_16x16x32_f16(false, afrag, false, bfrag,
                                                 (short)0, acc, false, false);
  }

  const int cout = nSub * 16 + (lane & 15);
  const float bias = b1[cout];
  const size_t rowBase = (size_t)b * HWPX + pixBase + mSub * 16 + (laneHi ? 8 : 0);
#pragma unroll
  for (int r = 0; r < 8; ++r)
    tokens[(rowBase + r) * CC + cout] = (_Float16)(acc[r] + bias);
}

// ---------------------------------------------------------------------------
// Fused grid-sample + patch GEMM.
//   block = 256 thr (8 waves). M-tile = 16 patches, N-tile = 256 couts.
//   Phase 1: each thread bilinear-samples one (patch,pixel) channel vector
//            (64 f16) into a shared A tile [16][1024].
//   Phase 2: each wave owns a 16x32 output tile: per K-chunk one shared
//            A fragment feeds two independent wmma chains (2 B fragments).
// ---------------------------------------------------------------------------
#define AROW 1040   // f16 row stride: 2080B -> 16B aligned, bank skew

__global__ void sample_gemm_kernel(const _Float16* __restrict__ tokens,
                                   const float* __restrict__ grid,
                                   const _Float16* __restrict__ w2p,
                                   const float* __restrict__ b2,
                                   float* __restrict__ out) {
  __shared__ _Float16 Atile[16 * AROW];

  const int mBase = blockIdx.x * 16;
  const int b = mBase / PPI;                        // tile never crosses b
  const int t = threadIdx.x;

  // ---- Phase 1: gather 16 patches x 16 pixels, bilinear, zeros padding ----
  {
    const int patchLocal = t >> 4;
    const int pix = t & 15;
    const int m = mBase + patchLocal;
    const int rem = m - b * PPI;
    const int ph = rem / PW, pw = rem - (rem / PW) * PW;
    const int dy = pix >> 2, dx = pix & 3;
    const int h = ph * PP + dy, w = pw * PP + dx;

    const size_t gi = ((size_t)(b * HH + h) * WW + w) * 2;
    const float gx = grid[gi + 0];
    const float gy = grid[gi + 1];
    const float ix = (gx + 1.f) * 0.5f * (float)(WW - 1);
    const float iy = (gy + 1.f) * 0.5f * (float)(HH - 1);
    const float x0f = floorf(ix), y0f = floorf(iy);
    const float fx = ix - x0f, fy = iy - y0f;
    const int x0 = (int)x0f, y0 = (int)y0f;
    const int x1 = x0 + 1, y1 = y0 + 1;
    const float vx0 = (x0 >= 0 && x0 < WW) ? 1.f : 0.f;
    const float vx1 = (x1 >= 0 && x1 < WW) ? 1.f : 0.f;
    const float vy0 = (y0 >= 0 && y0 < HH) ? 1.f : 0.f;
    const float vy1 = (y1 >= 0 && y1 < HH) ? 1.f : 0.f;
    const int x0c = min(max(x0, 0), WW - 1), x1c = min(max(x1, 0), WW - 1);
    const int y0c = min(max(y0, 0), HH - 1), y1c = min(max(y1, 0), HH - 1);
    const float w00 = (1.f - fx) * (1.f - fy) * vx0 * vy0;
    const float w10 = fx * (1.f - fy) * vx1 * vy0;
    const float w01 = (1.f - fx) * fy * vx0 * vy1;
    const float w11 = fx * fy * vx1 * vy1;

    const size_t ibase = (size_t)b * HWPX;
    const _Float16* p00 = tokens + (ibase + (size_t)y0c * WW + x0c) * CC;
    const _Float16* p10 = tokens + (ibase + (size_t)y0c * WW + x1c) * CC;
    const _Float16* p01 = tokens + (ibase + (size_t)y1c * WW + x0c) * CC;
    const _Float16* p11 = tokens + (ibase + (size_t)y1c * WW + x1c) * CC;

    _Float16* dstRow = &Atile[patchLocal * AROW + pix * CC];
#pragma unroll
    for (int cc = 0; cc < 8; ++cc) {
      const int o = cc * 8;
      v8h t00 = *(const v8h*)(p00 + o);
      v8h t10 = *(const v8h*)(p10 + o);
      v8h t01 = *(const v8h*)(p01 + o);
      v8h t11 = *(const v8h*)(p11 + o);
      v8h d;
#pragma unroll
      for (int i = 0; i < 8; ++i) {
        float v = w00 * (float)t00[i] + w10 * (float)t10[i] +
                  w01 * (float)t01[i] + w11 * (float)t11[i];
        d[i] = (_Float16)v;
      }
      *(v8h*)(dstRow + o) = d;
    }
  }
  __syncthreads();

  // ---- Phase 2: per-wave 16x32 tile, K = 1024 in 32 steps x 2 wmma ----
  const int wave = t >> 5;
  const int lane = t & 31;
  const int laneHi = lane >= 16;
  const int eWave = blockIdx.y * 256 + wave * 32;

  const _Float16* aRow  = &Atile[(lane & 15) * AROW];
  const _Float16* bRow0 = w2p + (size_t)(eWave + (lane & 15)) * KTOT;
  const _Float16* bRow1 = bRow0 + (size_t)16 * KTOT;

  v8f acc0 = {};
  v8f acc1 = {};
#pragma unroll 4
  for (int kc = 0; kc < 32; ++kc) {
    const int kbase = kc * 32;
    v16h afrag = load_fragA_k32(aRow, kbase, laneHi);
    v16h bf0   = load_fragB_k32(bRow0, kbase, laneHi);
    v16h bf1   = load_fragB_k32(bRow1, kbase, laneHi);
    acc0 = __builtin_amdgcn_wmma_f32_16x16x32_f16(false, afrag, false, bf0,
                                                  (short)0, acc0, false, false);
    acc1 = __builtin_amdgcn_wmma_f32_16x16x32_f16(false, afrag, false, bf1,
                                                  (short)0, acc1, false, false);
  }

  const int e0 = eWave + (lane & 15);
  const int e1 = e0 + 16;
  const float bias0 = b2[e0];
  const float bias1 = b2[e1];
  const int remBase = mBase - b * PPI + (laneHi ? 8 : 0);
  float* out0 = out + (size_t)b * (EE * PPI) + (size_t)e0 * PPI + remBase;
  float* out1 = out + (size_t)b * (EE * PPI) + (size_t)e1 * PPI + remBase;
#pragma unroll
  for (int r = 0; r < 8; ++r) out0[r] = acc0[r] + bias0;
#pragma unroll
  for (int r = 0; r < 8; ++r) out1[r] = acc1[r] + bias1;
}

// ---------------------------------------------------------------------------
// Launcher. Inputs: x, w1, b1, w2, b2, w_off, b_off, w_sc, b_sc (f32).
// Workspace: tokens f16 (51.4MB) | grid f32 (3.2MB) | w1p f16 (8KB) | w2p f16 (1.5MB)
// ---------------------------------------------------------------------------
extern "C" void kernel_launch(void* const* d_in, const int* in_sizes, int n_in,
                              void* d_out, int out_size, void* d_ws,
                              size_t ws_size, hipStream_t stream) {
  const float* x     = (const float*)d_in[0];
  const float* w1    = (const float*)d_in[1];
  const float* b1    = (const float*)d_in[2];
  const float* w2    = (const float*)d_in[3];
  const float* b2    = (const float*)d_in[4];
  const float* w_off = (const float*)d_in[5];
  const float* b_off = (const float*)d_in[6];
  const float* w_sc  = (const float*)d_in[7];
  const float* b_sc  = (const float*)d_in[8];
  float* out = (float*)d_out;

  char* ws = (char*)d_ws;
  const size_t tokensBytes = (size_t)BB * HWPX * CC * sizeof(_Float16);
  const size_t gridBytes   = (size_t)BB * HWPX * 2 * sizeof(float);
  _Float16* tokens = (_Float16*)ws;
  float*    gridWS = (float*)(ws + tokensBytes);
  _Float16* w1p    = (_Float16*)(ws + tokensBytes + gridBytes);
  _Float16* w2p    = (_Float16*)(ws + tokensBytes + gridBytes +
                                 (size_t)CC * CC * sizeof(_Float16));

  repack_w1_kernel<<<(CC * CC + 255) / 256, 256, 0, stream>>>(w1, w1p);
  repack_w2_kernel<<<(EE * KTOT + 255) / 256, 256, 0, stream>>>(w2, w2p);

  patch_grid_kernel<<<NPATCH / 8, 256, 0, stream>>>(x, w_off, b_off, w_sc,
                                                    b_sc, gridWS);

  tokens_kernel<<<BB * (HWPX / 32), 256, 0, stream>>>(x, w1p, b1, tokens);

  dim3 g(NPATCH / 16, EE / 256);
  sample_gemm_kernel<<<g, 256, 0, stream>>>(tokens, gridWS, w2p, b2, out);
}
